// FIXED_POINT_QUANT_SIMPLE_BED_DETECTOR_10273561772343
// MI455X (gfx1250) — 8-bit matrix operand order repaired and checked by reading, compiled, not run on hardware
//
#include <hip/hip_runtime.h>
#include <cstdint>

typedef int v8i __attribute__((ext_vector_type(8)));

__device__ __forceinline__ float pow2ceil(float t) {
  t = fmaxf(t, 1e-8f);
  return exp2f(ceilf(log2f(t)));
}

__device__ __forceinline__ void wave_fence() {
  asm volatile("" ::: "memory");
  __builtin_amdgcn_wave_barrier();
}

__global__ void zero_f32(float* p, int n) {
  int i = blockIdx.x * blockDim.x + threadIdx.x;
  if (i < n) p[i] = 0.0f;
}

__global__ void quant_input(const float* __restrict__ x, uint8_t* __restrict__ out,
                            int total) {
  int idx = blockIdx.x * blockDim.x + threadIdx.x;
  if (idx >= total) return;
  int c = idx % 3; int r = idx / 3;
  int w = r % 224; r /= 224;
  int h = r % 224; int img = r / 224;
  float v = x[(((size_t)img * 3 + c) * 224 + h) * 224 + w];
  float q = fminf(fmaxf(rintf(v * 256.0f), 0.0f), 255.0f);
  out[idx] = (uint8_t)q;
}

__global__ void wquant(const float* __restrict__ w, int8_t* __restrict__ wqout,
                       float* __restrict__ wscale,
                       int Cin, int Cout, int ksz, int Kreal, int Kpad) {
  int n = blockIdx.x;
  int tid = threadIdx.x;
  __shared__ float red[256];
  __shared__ float s_scale;
  float m = 0.0f;
  if (n < Cout)
    for (int t = tid; t < Kreal; t += 256)
      m = fmaxf(m, fabsf(w[(size_t)n * Kreal + t]));
  red[tid] = m;
  __syncthreads();
  for (int s = 128; s > 0; s >>= 1) {
    if (tid < s) red[tid] = fmaxf(red[tid], red[tid + s]);
    __syncthreads();
  }
  if (tid == 0) {
    float sc = (n < Cout) ? (pow2ceil(red[0]) * (1.0f / 128.0f)) : 1.0f;
    s_scale = sc;
    wscale[n] = sc;
  }
  __syncthreads();
  float inv = 1.0f / s_scale;
  for (int t = tid; t < Kpad; t += 256) {
    signed char q = 0;
    if (n < Cout && t < Kreal) {
      int p = t / Cin;
      int ci = t - p * Cin;
      int dy = p / ksz;
      int dx = p - dy * ksz;
      float val = w[(size_t)n * Kreal + ((size_t)ci * ksz + dy) * ksz + dx];
      float r = fminf(fmaxf(rintf(val * inv), -128.0f), 127.0f);
      q = (signed char)r;
    }
    wqout[(size_t)n * Kpad + t] = q;
  }
}

__global__ void bquant(const float* __restrict__ b, float* __restrict__ bqout, int Cout) {
  __shared__ float red[64];
  int t = threadIdx.x;
  float v = (t < Cout) ? b[t] : 0.0f;
  red[t] = fabsf(v);
  __syncthreads();
  for (int s = 32; s > 0; s >>= 1) {
    if (t < s) red[t] = fmaxf(red[t], red[t + s]);
    __syncthreads();
  }
  float scale = pow2ceil(red[0]) * (1.0f / 128.0f);
  float q = fminf(fmaxf(rintf(v / scale), -128.0f), 127.0f) * scale;
  bqout[t] = (t < Cout) ? q : 0.0f;
}

template <int CIN, int KSZ, int NCHUNK, int NT>
__global__ void __launch_bounds__(256)
conv_wmma_iu8(const uint8_t* __restrict__ act,
              const int8_t* __restrict__ wq,
              const float* __restrict__ wscale,
              const float* __restrict__ bq,
              float* __restrict__ outf,
              float* __restrict__ layer_max,
              const float* __restrict__ prev_max,
              float const_scale,
              int H, int Cout, int do_relu, int M) {
  constexpr int KREAL = CIN * KSZ * KSZ;
  constexpr int KPAD = NCHUNK * 64;
  constexpr int PAD = KSZ / 2;
  constexpr int RS = 80;

  __shared__ uint8_t sA[8][16 * RS];

  const int tid = threadIdx.x;
  const int wv = tid >> 5;
  const int lane = tid & 31;
  const int row = lane & 15;
  const int hh = lane >> 4;

  const float aScale =
      prev_max ? (pow2ceil(*prev_max) * (1.0f / 256.0f)) : const_scale;

  const int m0 = (blockIdx.x * 8 + wv) * 16;
  const int n0 = blockIdx.y * NT * 16;

  const int m = m0 + row;
  const bool mvalid = (m < M);
  int img = 0, oh = 0, ow = 0;
  if (mvalid) {
    img = m / (H * H);
    int rem = m - img * H * H;
    oh = rem / H;
    ow = rem - oh * H;
  }

  v8i acc[NT];
#pragma unroll
  for (int t = 0; t < NT; ++t)
#pragma unroll
    for (int i = 0; i < 8; ++i) acc[t][i] = 0;

  const int8_t* wp_base[NT];
#pragma unroll
  for (int t = 0; t < NT; ++t)
    wp_base[t] = wq + (size_t)(n0 + t * 16 + row) * KPAD + hh * 16;

#pragma unroll
  for (int c = 0; c < NCHUNK; ++c) {
    const int kc = c * 64;

    if constexpr (CIN % 16 == 0) {
#pragma unroll
      for (int s = 0; s < 2; ++s) {
        const int k0 = kc + hh * 32 + s * 16;
        uint4 v = make_uint4(0u, 0u, 0u, 0u);
        if (k0 < KREAL && mvalid) {
          int p = k0 / CIN;
          int ci = k0 % CIN;
          int dy = p / KSZ;
          int dx = p - dy * KSZ;
          int ih = oh + dy - PAD;
          int iw = ow + dx - PAD;
          if (ih >= 0 && ih < H && iw >= 0 && iw < H)
            v = *(const uint4*)(act +
                                (((size_t)img * H + ih) * H + iw) * CIN + ci);
        }
        *(uint4*)(&sA[wv][row * RS + hh * 32 + s * 16]) = v;
      }
    } else {
      uint8_t* dst = &sA[wv][row * RS + hh * 32];
#pragma unroll
      for (int g = 0; g < 8; ++g) {
        unsigned packed = 0;
#pragma unroll
        for (int b = 0; b < 4; ++b) {
          int k = kc + hh * 32 + g * 4 + b;
          unsigned v = 0;
          if (mvalid && k < KREAL) {
            int p = k / CIN;
            int ci = k - p * CIN;
            int dy = p / KSZ;
            int dx = p - dy * KSZ;
            int ih = oh + dy - PAD;
            int iw = ow + dx - PAD;
            if (ih >= 0 && ih < H && iw >= 0 && iw < H)
              v = act[(((size_t)img * H + ih) * H + iw) * CIN + ci];
          }
          packed |= v << (8 * b);
        }
        *(unsigned*)(dst + g * 4) = packed;
      }
    }
    wave_fence();

    const uint8_t* ap = &sA[wv][row * RS + hh * 16];
    uint2 q0 = *(const uint2*)(ap);
    uint2 q1 = *(const uint2*)(ap + 8);
    uint2 q2 = *(const uint2*)(ap + 32);
    uint2 q3 = *(const uint2*)(ap + 40);
    v8i a;
    a[0] = (int)q0.x; a[1] = (int)q0.y;
    a[2] = (int)q1.x; a[3] = (int)q1.y;
    a[4] = (int)q2.x; a[5] = (int)q2.y;
    a[6] = (int)q3.x; a[7] = (int)q3.y;

#pragma unroll
    for (int t = 0; t < NT; ++t) {
      const int8_t* wp = wp_base[t] + kc;
      int4 b0 = *(const int4*)(wp);
      int4 b1 = *(const int4*)(wp + 32);
      v8i bb;
      bb[0] = b0.x; bb[1] = b0.y; bb[2] = b0.z; bb[3] = b0.w;
      bb[4] = b1.x; bb[5] = b1.y; bb[6] = b1.z; bb[7] = b1.w;

      if (kc + 64 < KPAD) __builtin_prefetch(wp + 64, 0, 1);

      acc[t] = __builtin_amdgcn_wmma_i32_16x16x64_iu8(false, a, true, bb,
                                                      acc[t], false, false);
    }
    wave_fence();
  }

  float lmax = 0.0f;
#pragma unroll
  for (int t = 0; t < NT; ++t) {
    const int n = n0 + t * 16 + row;
    const float sAB = aScale * wscale[n];
    const float bv = bq[n];
#pragma unroll
    for (int i = 0; i < 8; ++i) {
      int mr = m0 + i + 8 * hh;
      if (mr < M && n < Cout) {
        float v = (float)acc[t][i] * sAB + bv;
        if (do_relu) v = fmaxf(v, 0.0f);
        outf[(size_t)mr * Cout + n] = v;
        lmax = fmaxf(lmax, v);
      }
    }
  }
  if (do_relu) {
#pragma unroll
    for (int off = 16; off > 0; off >>= 1)
      lmax = fmaxf(lmax, __shfl_xor(lmax, off, 32));
    if (lane == 0) atomicMax((int*)layer_max, __float_as_int(lmax));
  }
}

__global__ void actquant(const float* __restrict__ f, uint8_t* __restrict__ out,
                         const float* __restrict__ maxptr,
                         int H, int C, int pool, int Hp, int total) {
  int idx = blockIdx.x * blockDim.x + threadIdx.x;
  if (idx >= total) return;
  float scale = pow2ceil(*maxptr) * (1.0f / 256.0f);
  float inv = 1.0f / scale;
  int c = idx % C; int r = idx / C;
  int ow = r % Hp; r /= Hp;
  int oh = r % Hp; int img = r / Hp;
  float v;
  if (pool) {
    const float* base = f + (((size_t)img * H + oh * 2) * H + ow * 2) * C + c;
    size_t rs = (size_t)H * C;
    v = fmaxf(fmaxf(base[0], base[C]), fmaxf(base[rs], base[rs + C]));
  } else {
    v = f[(((size_t)img * H + oh) * H + ow) * C + c];
  }
  float q = fminf(fmaxf(rintf(v * inv), 0.0f), 255.0f);
  out[idx] = (uint8_t)q;
}

__global__ void to_nchw(const float* __restrict__ f, float* __restrict__ out) {
  int idx = blockIdx.x * blockDim.x + threadIdx.x;
  if (idx >= 32 * 11 * 7 * 7) return;
  int ow = idx % 7; int r = idx / 7;
  int oh = r % 7; r /= 7;
  int c = r % 11; int img = r / 11;
  out[idx] = f[(((size_t)img * 7 + oh) * 7 + ow) * 11 + c];
}

struct LayerCfg { int cin, cout, k, H, pool, relu; };
static const LayerCfg LAYERS[24] = {
    {3, 32, 3, 224, 1, 1},  {32, 16, 3, 112, 1, 1}, {16, 16, 1, 56, 0, 1},
    {16, 32, 3, 56, 0, 1},  {32, 32, 1, 56, 0, 1},  {32, 64, 3, 56, 1, 1},
    {64, 32, 1, 28, 0, 1},  {32, 64, 3, 28, 0, 1},  {64, 32, 1, 28, 0, 1},
    {32, 64, 3, 28, 0, 1},  {64, 32, 1, 28, 0, 1},  {32, 64, 3, 28, 1, 1},
    {64, 32, 1, 14, 0, 1},  {32, 64, 3, 14, 0, 1},  {64, 32, 1, 14, 0, 1},
    {32, 64, 3, 14, 0, 1},  {64, 64, 3, 14, 0, 1},  {64, 64, 3, 14, 1, 1},
    {64, 64, 3, 7, 0, 1},   {64, 64, 3, 7, 0, 1},   {64, 64, 1, 7, 0, 1},
    {64, 16, 1, 7, 0, 1},   {16, 16, 1, 7, 0, 1},   {16, 11, 1, 7, 0, 0},
};

extern "C" void kernel_launch(void* const* d_in, const int* in_sizes, int n_in,
                              void* d_out, int out_size, void* d_ws, size_t ws_size,
                              hipStream_t stream) {
  (void)in_sizes; (void)n_in; (void)out_size; (void)ws_size;
  const float* x = (const float*)d_in[0];

  char* ws = (char*)d_ws;
  const size_t ACT_BYTES = (size_t)16 << 20;
  const size_t F32_BYTES = (size_t)32 * 224 * 224 * 32 * 4;
  float* maxvals = (float*)ws;
  uint8_t* actA = (uint8_t*)(ws + 256);
  uint8_t* actB = (uint8_t*)(ws + 256 + ACT_BYTES);
  float* f32buf = (float*)(ws + 256 + 2 * ACT_BYTES);
  int8_t* wqbuf = (int8_t*)(ws + 256 + 2 * ACT_BYTES + F32_BYTES);
  float* wscale = (float*)(ws + 256 + 2 * ACT_BYTES + F32_BYTES + 65536);
  float* bqbuf = wscale + 64;

  zero_f32<<<1, 32, 0, stream>>>(maxvals, 24);

  int in_total = 32 * 3 * 224 * 224;
  quant_input<<<(in_total + 255) / 256, 256, 0, stream>>>(x, actA, in_total);

  uint8_t* cur = actA;
  uint8_t* nxt = actB;
  for (int i = 0; i < 24; ++i) {
    const LayerCfg L = LAYERS[i];
    const int Kreal = L.cin * L.k * L.k;
    const int Kpad = ((Kreal + 63) / 64) * 64;
    const int CoutPad = ((L.cout + 15) / 16) * 16;
    const int M = 32 * L.H * L.H;

    wquant<<<CoutPad, 256, 0, stream>>>((const float*)d_in[1 + 2 * i], wqbuf,
                                        wscale, L.cin, L.cout, L.k, Kreal, Kpad);
    bquant<<<1, 64, 0, stream>>>((const float*)d_in[2 + 2 * i], bqbuf, L.cout);

#define LAUNCH_CONV(CI, KS, NC, NTILES)                                       \
  do {                                                                        \
    dim3 cg((M + 127) / 128, CoutPad / (16 * (NTILES)));                      \
    conv_wmma_iu8<CI, KS, NC, NTILES><<<cg, 256, 0, stream>>>(                \
        cur, wqbuf, wscale, bqbuf, f32buf, maxvals + i,                       \
        (i == 0) ? nullptr : (maxvals + i - 1), 1.0f / 256.0f, L.H, L.cout,   \
        L.relu, M);                                                           \
  } while (0)
    switch (L.cin * 1000 + L.k * 100 + CoutPad) {
      case 3332:  LAUNCH_CONV(3, 3, 1, 2);  break;
      case 32316: LAUNCH_CONV(32, 3, 5, 1); break;
      case 16116: LAUNCH_CONV(16, 1, 1, 1); break;
      case 16332: LAUNCH_CONV(16, 3, 3, 2); break;
      case 32132: LAUNCH_CONV(32, 1, 1, 2); break;
      case 32364: LAUNCH_CONV(32, 3, 5, 4); break;
      case 64132: LAUNCH_CONV(64, 1, 1, 2); break;
      case 64364: LAUNCH_CONV(64, 3, 9, 4); break;
      case 64164: LAUNCH_CONV(64, 1, 1, 4); break;
      case 64116: LAUNCH_CONV(64, 1, 1, 1); break;
    }
#undef LAUNCH_CONV

    if (i < 23) {
      int Hp = L.pool ? (L.H / 2) : L.H;
      int tot = 32 * Hp * Hp * L.cout;
      actquant<<<(tot + 255) / 256, 256, 0, stream>>>(f32buf, nxt, maxvals + i,
                                                      L.H, L.cout, L.pool, Hp, tot);
      uint8_t* t = cur; cur = nxt; nxt = t;
    } else {
      to_nchw<<<(32 * 11 * 49 + 255) / 256, 256, 0, stream>>>(f32buf, (float*)d_out);
    }
  }
}
